// AttentionPool_26259430048673
// MI455X (gfx1250) — matrix operand order repaired and checked by reading, not compiled, not run on hardware
//
#include <hip/hip_runtime.h>
#include <hip/hip_bf16.h>
#include <math.h>

// Problem constants (from reference): B=32, S=2048, D=1024
#define BATCH 32
#define SEQ   2048
#define DDIM  1024
#define ROWS  (BATCH * SEQ)        // 65536

// GEMM tiling for the score kernel
#define BMROWS 128                 // rows per block
#define BN     64                  // N tile width (columns of W1 per outer step)
#define BK     64                  // K tile depth
#define KPAD   72                  // LDS row stride in bf16 elems (144B, 16B-aligned)
#define NKT    (DDIM / BK)         // 16 K tiles
#define NNT    (DDIM / BN)         // 16 N tiles
#define PER_TILE 6                 // async b128 ops issued per thread per tile

typedef __bf16 bf16x16 __attribute__((ext_vector_type(16)));
typedef float  floatx8 __attribute__((ext_vector_type(8)));
typedef int    i32x4   __attribute__((ext_vector_type(4)));

union Frag16 {
    bf16x16 f;
    uint4   q[2];
};

// --- CDNA5 async global->LDS staging (guarded; sync fallback keeps compile safe)
#if defined(__HIP_DEVICE_COMPILE__) &&                                        \
    __has_builtin(__builtin_amdgcn_global_load_async_to_lds_b128) &&          \
    __has_builtin(__builtin_amdgcn_s_wait_asynccnt)
#define USE_ASYNC 1
#else
#define USE_ASYNC 0
#endif

#if USE_ASYNC
typedef __attribute__((address_space(1))) i32x4 gvec_t;   // global int4
typedef __attribute__((address_space(3))) i32x4 lvec_t;   // LDS int4
#define WAIT_ASYNC(n) __builtin_amdgcn_s_wait_asynccnt(n)
#else
#define WAIT_ASYNC(n) ((void)0)
#endif

__device__ __forceinline__ void cp_async16(const unsigned short* g,
                                           unsigned short* l) {
#if USE_ASYNC
    __builtin_amdgcn_global_load_async_to_lds_b128(
        (gvec_t*)g, (lvec_t*)l, 0, 0);
#else
    *(uint4*)l = *(const uint4*)g;
#endif
}

__device__ inline unsigned short f2bf(float f) {
    unsigned u = __builtin_bit_cast(unsigned, f);
    unsigned r = u + 0x7FFFu + ((u >> 16) & 1u);   // round-to-nearest-even
    return (unsigned short)(r >> 16);
}

// ---------------------------------------------------------------------------
// K0: W1 [D][D] fp32 -> W1t [e][d] bf16 (transposed so B-fragments read
// contiguous K per lane)
// ---------------------------------------------------------------------------
__global__ __launch_bounds__(256) void convert_w1_kernel(
    const float* __restrict__ W1, unsigned short* __restrict__ w1t) {
    int i = blockIdx.x * 256 + threadIdx.x;        // i = e*D + d
    int e = i >> 10;
    int d = i & (DDIM - 1);
    w1t[i] = f2bf(W1[(size_t)d * DDIM + e]);
}

// ---------------------------------------------------------------------------
// K1: LayerNorm each row of x, emit bf16 normalized rows. 1 block / row.
// ---------------------------------------------------------------------------
__global__ __launch_bounds__(256) void ln_kernel(
    const float* __restrict__ x, const float* __restrict__ gamma,
    const float* __restrict__ beta, unsigned short* __restrict__ xnb) {
    const int t = threadIdx.x;
    const size_t row = blockIdx.x;
    const float4 v = ((const float4*)(x + row * DDIM))[t];

    float s  = v.x + v.y + v.z + v.w;
    float ss = v.x * v.x + v.y * v.y + v.z * v.z + v.w * v.w;
    #pragma unroll
    for (int o = 16; o > 0; o >>= 1) {
        s  += __shfl_down(s, o, 32);
        ss += __shfl_down(ss, o, 32);
    }
    __shared__ float rbuf[16];
    __shared__ float stats[2];
    if ((t & 31) == 0) { rbuf[t >> 5] = s; rbuf[8 + (t >> 5)] = ss; }
    __syncthreads();
    if (t == 0) {
        float a = 0.f, b = 0.f;
        #pragma unroll
        for (int i = 0; i < 8; ++i) { a += rbuf[i]; b += rbuf[8 + i]; }
        float mean = a * (1.0f / DDIM);
        float var  = b * (1.0f / DDIM) - mean * mean;
        stats[0] = mean;
        stats[1] = rsqrtf(var + 1e-5f);
    }
    __syncthreads();
    const float mean = stats[0], rstd = stats[1];

    const float4 g4 = ((const float4*)gamma)[t];
    const float4 b4 = ((const float4*)beta)[t];
    float y0 = (v.x - mean) * rstd * g4.x + b4.x;
    float y1 = (v.y - mean) * rstd * g4.y + b4.y;
    float y2 = (v.z - mean) * rstd * g4.z + b4.z;
    float y3 = (v.w - mean) * rstd * g4.w + b4.w;
    uint2 p;
    p.x = (unsigned)f2bf(y0) | ((unsigned)f2bf(y1) << 16);
    p.y = (unsigned)f2bf(y2) | ((unsigned)f2bf(y3) << 16);
    ((uint2*)(xnb + row * DDIM))[t] = p;
}

// ---------------------------------------------------------------------------
// K2: scores[r] = sum_n tanh( (xn @ W1)[r,n] + b1[n] ) * W2[n]
// bf16 WMMA GEMM, async-to-LDS double-buffered tiles, fused tanh/W2 epilogue.
// 8 waves as 4(M) x 2(N); each wave owns a 32x32 C tile (4 v8f accumulators).
// ---------------------------------------------------------------------------
__global__ __launch_bounds__(256) void score_kernel(
    const unsigned short* __restrict__ xnb,   // [ROWS][D] bf16
    const unsigned short* __restrict__ w1t,   // [D(e)][D(d)] bf16 (transposed)
    const float* __restrict__ b1,
    const float* __restrict__ w2,
    float* __restrict__ scores) {
    __shared__ __align__(16) unsigned short As[2][BMROWS][KPAD];
    __shared__ __align__(16) unsigned short Bs[2][BN][KPAD];
    __shared__ float b1s[DDIM];
    __shared__ float w2s[DDIM];
    __shared__ float ssum[BMROWS];

    const int t    = threadIdx.x;
    const int lane = t & 31;
    const int wid  = t >> 5;
    const int wm   = wid & 3;          // 0..3 -> 32-row M slab
    const int wn   = wid >> 2;         // 0..1 -> 32-col N slab
    const int lr   = lane & 15;
    const bool hi  = lane >= 16;
    const int m0a  = wm * 32;          // first 16-row subtile
    const int m0b  = m0a + 16;         // second 16-row subtile
    const int n0w  = wn * 32;
    const size_t rowBase = (size_t)blockIdx.x * BMROWS;

    for (int i = t; i < DDIM; i += 256) { b1s[i] = b1[i]; w2s[i] = w2[i]; }
    for (int i = t; i < BMROWS; i += 256) ssum[i] = 0.f;

    // Cooperative tile-staging map.
    const int ar = t >> 1;             // A row 0..127
    const int ac = (t & 1) * 32;       // A col segment (4 x b128)
    const int br = t >> 2;             // B row 0..63
    const int bc = (t & 3) * 16;       // B col segment (2 x b128)

    for (int nt = 0; nt < NNT; ++nt) {
        floatx8 c00 = {0.f,0.f,0.f,0.f,0.f,0.f,0.f,0.f};
        floatx8 c01 = {0.f,0.f,0.f,0.f,0.f,0.f,0.f,0.f};
        floatx8 c10 = {0.f,0.f,0.f,0.f,0.f,0.f,0.f,0.f};
        floatx8 c11 = {0.f,0.f,0.f,0.f,0.f,0.f,0.f,0.f};

        // Prologue: stage tile kt=0 into buffer 0.
        {
            const unsigned short* gA = xnb + (rowBase + ar) * DDIM + ac;
            const unsigned short* gB = w1t + ((size_t)nt * BN + br) * DDIM + bc;
            #pragma unroll
            for (int j = 0; j < 4; ++j) cp_async16(gA + j * 8, &As[0][ar][ac + j * 8]);
            #pragma unroll
            for (int j = 0; j < 2; ++j) cp_async16(gB + j * 8, &Bs[0][br][bc + j * 8]);
        }

        for (int kt = 0; kt < NKT; ++kt) {
            const int cur = kt & 1;
            if (kt + 1 < NKT) {        // stage next tile into the other buffer
                const int nb = cur ^ 1;
                const unsigned short* gA =
                    xnb + (rowBase + ar) * DDIM + (kt + 1) * BK + ac;
                const unsigned short* gB =
                    w1t + ((size_t)nt * BN + br) * DDIM + (kt + 1) * BK + bc;
                #pragma unroll
                for (int j = 0; j < 4; ++j)
                    cp_async16(gA + j * 8, &As[nb][ar][ac + j * 8]);
                #pragma unroll
                for (int j = 0; j < 2; ++j)
                    cp_async16(gB + j * 8, &Bs[nb][br][bc + j * 8]);
                WAIT_ASYNC(PER_TILE);  // tile kt complete (in-order), kt+1 in flight
            } else {
                WAIT_ASYNC(0);
            }
            __syncthreads();           // tile kt visible to all waves

            #pragma unroll
            for (int ks = 0; ks < BK; ks += 32) {
                // A fragment, ISA 16-bit 16x32 layout:
                // lanes 0-15 : M=lr, K = ks+[0..7] then ks+[16..23]
                // lanes 16-31: M=lr, K = ks+[8..15] then ks+[24..31]
                const int ao = ks + (hi ? 8 : 0);
                Frag16 a0, a1;
                a0.q[0] = *(const uint4*)&As[cur][m0a + lr][ao];
                a0.q[1] = *(const uint4*)&As[cur][m0a + lr][ao + 16];
                a1.q[0] = *(const uint4*)&As[cur][m0b + lr][ao];
                a1.q[1] = *(const uint4*)&As[cur][m0b + lr][ao + 16];
                // B fragment, ISA 16-bit 32x16 layout:
                // lanes 0-15 : N=lr, K = ks+[0..7] then ks+[16..23]; lanes 16-31: K = ks+[8..15] then ks+[24..31]
                const int bo = ks + (hi ? 8 : 0);
                Frag16 fb0, fb1;
                {
                    const uint4* p = (const uint4*)&Bs[cur][n0w + lr][bo];
                    fb0.q[0] = p[0]; fb0.q[1] = p[2];
                }
                {
                    const uint4* p = (const uint4*)&Bs[cur][n0w + 16 + lr][bo];
                    fb1.q[0] = p[0]; fb1.q[1] = p[2];
                }
                c00 = __builtin_amdgcn_wmma_f32_16x16x32_bf16(
                    false, a0.f, false, fb0.f, (short)0, c00, false, false);
                c01 = __builtin_amdgcn_wmma_f32_16x16x32_bf16(
                    false, a0.f, false, fb1.f, (short)0, c01, false, false);
                c10 = __builtin_amdgcn_wmma_f32_16x16x32_bf16(
                    false, a1.f, false, fb0.f, (short)0, c10, false, false);
                c11 = __builtin_amdgcn_wmma_f32_16x16x32_bf16(
                    false, a1.f, false, fb1.f, (short)0, c11, false, false);
            }
            __syncthreads();           // compute done before buffer reuse
        }

        // Fused epilogue: tanh(+b1) then rank-1 dot with W2, accumulated per row.
        // C layout: VGPR e -> M = m? + e (+8 for lanes 16-31), N = nt*BN+n0w+lr.
        #pragma unroll
        for (int e = 0; e < 8; ++e) {
            int rowA = m0a + e + (hi ? 8 : 0);
            int rowB = m0b + e + (hi ? 8 : 0);
            int col0 = nt * BN + n0w + lr;
            int col1 = col0 + 16;
            float w20 = w2s[col0], w21 = w2s[col1];
            float vA = tanhf(c00[e] + b1s[col0]) * w20 +
                       tanhf(c01[e] + b1s[col1]) * w21;
            float vB = tanhf(c10[e] + b1s[col0]) * w20 +
                       tanhf(c11[e] + b1s[col1]) * w21;
            atomicAdd(&ssum[rowA], vA);
            atomicAdd(&ssum[rowB], vB);
        }
    }
    __syncthreads();
    if (t < BMROWS) scores[rowBase + t] = ssum[t];
}

// ---------------------------------------------------------------------------
// K3: softmax over S per batch (b2 cancels inside softmax, skipped).
// ---------------------------------------------------------------------------
__global__ __launch_bounds__(256) void softmax_kernel(
    const float* __restrict__ scores, float* __restrict__ weights) {
    const int b = blockIdx.x;
    const int t = threadIdx.x;
    const float* s = scores + (size_t)b * SEQ;
    float loc[8];
    float mx = -INFINITY;
    #pragma unroll
    for (int i = 0; i < 8; ++i) {
        loc[i] = s[t + i * 256];
        mx = fmaxf(mx, loc[i]);
    }
    __shared__ float red[256];
    red[t] = mx;
    __syncthreads();
    for (int o = 128; o > 0; o >>= 1) {
        if (t < o) red[t] = fmaxf(red[t], red[t + o]);
        __syncthreads();
    }
    mx = red[0];
    __syncthreads();
    float sum = 0.f;
    #pragma unroll
    for (int i = 0; i < 8; ++i) {
        loc[i] = __expf(loc[i] - mx);
        sum += loc[i];
    }
    red[t] = sum;
    __syncthreads();
    for (int o = 128; o > 0; o >>= 1) {
        if (t < o) red[t] += red[t + o];
        __syncthreads();
    }
    const float inv = 1.0f / red[0];
    #pragma unroll
    for (int i = 0; i < 8; ++i)
        weights[(size_t)b * SEQ + t + i * 256] = loc[i] * inv;
}

// ---------------------------------------------------------------------------
// K4/K5: out[b,d] = sum_s x[b,s,d] * w[b,s]  (two-stage, deterministic)
// ---------------------------------------------------------------------------
#define SCHUNK 16
#define SPER   (SEQ / SCHUNK)   // 128

__global__ __launch_bounds__(256) void pool_partial_kernel(
    const float* __restrict__ x, const float* __restrict__ weights,
    float* __restrict__ partials) {
    const int ch = blockIdx.x;   // 0..15
    const int b  = blockIdx.y;   // 0..31
    const int t  = threadIdx.x;
    __shared__ float wsh[SPER];
    if (t < SPER) wsh[t] = weights[(size_t)b * SEQ + ch * SPER + t];
    __syncthreads();
    const float4* xb =
        (const float4*)(x + ((size_t)b * SEQ + (size_t)ch * SPER) * DDIM);
    float ax = 0.f, ay = 0.f, az = 0.f, aw = 0.f;
    for (int s = 0; s < SPER; ++s) {
        if (s + 8 < SPER)
            __builtin_prefetch(&xb[(size_t)(s + 8) * (DDIM / 4) + t], 0, 1);
        float4 xv = xb[(size_t)s * (DDIM / 4) + t];
        float w = wsh[s];
        ax += xv.x * w; ay += xv.y * w; az += xv.z * w; aw += xv.w * w;
    }
    float4 r; r.x = ax; r.y = ay; r.z = az; r.w = aw;
    ((float4*)(partials + ((size_t)(b * SCHUNK + ch)) * DDIM))[t] = r;
}

__global__ __launch_bounds__(256) void pool_reduce_kernel(
    const float* __restrict__ partials, float* __restrict__ out) {
    const int b = blockIdx.x;
    const int t = threadIdx.x;
    float ax = 0.f, ay = 0.f, az = 0.f, aw = 0.f;
    for (int ch = 0; ch < SCHUNK; ++ch) {
        float4 p = ((const float4*)(partials + ((size_t)(b * SCHUNK + ch)) * DDIM))[t];
        ax += p.x; ay += p.y; az += p.z; aw += p.w;
    }
    float4 r; r.x = ax; r.y = ay; r.z = az; r.w = aw;
    ((float4*)(out + (size_t)b * DDIM))[t] = r;
}

// ---------------------------------------------------------------------------
extern "C" void kernel_launch(void* const* d_in, const int* in_sizes, int n_in,
                              void* d_out, int out_size, void* d_ws, size_t ws_size,
                              hipStream_t stream) {
    const float* x     = (const float*)d_in[0];   // [B,S,D]
    const float* gamma = (const float*)d_in[1];   // [D]
    const float* beta  = (const float*)d_in[2];   // [D]
    const float* W1    = (const float*)d_in[3];   // [D,D]
    const float* b1    = (const float*)d_in[4];   // [D]
    const float* W2    = (const float*)d_in[5];   // [D,1]
    // d_in[6] = b2: cancels in softmax, unused.
    float* out = (float*)d_out;                   // [B,D]

    char* ws = (char*)d_ws;
    unsigned short* xnb = (unsigned short*)ws;                 // ROWS*D bf16
    ws += (size_t)ROWS * DDIM * sizeof(unsigned short);
    unsigned short* w1t = (unsigned short*)ws;                 // D*D bf16
    ws += (size_t)DDIM * DDIM * sizeof(unsigned short);
    float* scores = (float*)ws;                                // ROWS f32
    ws += (size_t)ROWS * sizeof(float);
    float* weights = (float*)ws;                               // ROWS f32
    ws += (size_t)ROWS * sizeof(float);
    float* partials = (float*)ws;                              // B*SCHUNK*D f32

    convert_w1_kernel<<<(DDIM * DDIM) / 256, 256, 0, stream>>>(W1, w1t);
    ln_kernel<<<ROWS, 256, 0, stream>>>(x, gamma, beta, xnb);
    score_kernel<<<ROWS / BMROWS, 256, 0, stream>>>(xnb, w1t, b1, W2, scores);
    softmax_kernel<<<BATCH, 256, 0, stream>>>(scores, weights);
    pool_partial_kernel<<<dim3(SCHUNK, BATCH), 256, 0, stream>>>(x, weights, partials);
    pool_reduce_kernel<<<BATCH, 256, 0, stream>>>(partials, out);
}
